// SelfAttention_18863496364696
// MI455X (gfx1250) — matrix operand order repaired and checked by reading, not compiled, not run on hardware
//
#include <hip/hip_runtime.h>
#include <hip/hip_bf16.h>

// ---- CDNA5 WMMA types ----
typedef __attribute__((ext_vector_type(16))) __bf16 v16bf;
typedef __attribute__((ext_vector_type(8)))  __bf16 v8bf;
typedef __attribute__((ext_vector_type(8)))  float  v8f;

union F8 { v8f v; float f[8]; };

// fp32 -> bf16 round-to-nearest-even
static __device__ __forceinline__ __bf16 f2bf(float f) {
  union { float f; unsigned u; } a; a.f = f;
  unsigned r = a.u + 0x7fffu + ((a.u >> 16) & 1u);
  union { unsigned short s; __bf16 b; } o;
  o.s = (unsigned short)(r >> 16);
  return o.b;
}

static __device__ __forceinline__ v8f wmma_bf16(v16bf a, v16bf b, v8f c) {
  // (neg_a, A, neg_b, B, c_mod, C, reuse_a, reuse_b)
  return __builtin_amdgcn_wmma_f32_16x16x32_bf16(false, a, false, b, (short)0, c,
                                                 false, false);
}

// A fragment (16x32 bf16), A row-major with leading dim ldk.
// lane L: row = rowBase + L%16; halves[0..7]=K[k0+8*half ..], halves[8..15]=K[k0+16+8*half ..]
static __device__ __forceinline__ v16bf load_a_frag(const __bf16* A, int ldk,
                                                    int rowBase, int k0, int lane) {
  const int half = lane >> 4, lr = lane & 15;
  const __bf16* p = A + (size_t)(rowBase + lr) * ldk + k0 + 8 * half;
  union { v16bf v; v8bf h[2]; } u;
  u.h[0] = *(const v8bf*)p;
  u.h[1] = *(const v8bf*)(p + 16);
  return u.v;
}

// B fragment (32x16 bf16) from transposed B (BT is N x K row-major, leading dim ldk).
// lane L: N = colBase + L%16; halves[0..7]=K[k0+8*half ..], halves[8..15]=K[k0+16+8*half ..]
static __device__ __forceinline__ v16bf load_b_frag(const __bf16* BT, int ldk,
                                                    int colBase, int k0, int lane) {
  const int half = lane >> 4, lr = lane & 15;
  const __bf16* p = BT + (size_t)(colBase + lr) * ldk + k0 + 8 * half;
  union { v16bf v; v8bf h[2]; } u;
  u.h[0] = *(const v8bf*)p;
  u.h[1] = *(const v8bf*)(p + 16);
  return u.v;
}

// ---------------- pack kernels ----------------
__global__ void pack_f32_to_bf16(const float* __restrict__ x,
                                 __bf16* __restrict__ y, int n) {
  int i = blockIdx.x * blockDim.x + threadIdx.x;
  if (i < n) y[i] = f2bf(x[i]);
}

// W is K x N row-major; WT is N x K row-major (bf16)
__global__ void pack_transpose(const float* __restrict__ W,
                               __bf16* __restrict__ WT, int K, int N) {
  int i = blockIdx.x * blockDim.x + threadIdx.x;
  if (i >= N * K) return;
  int n = i / K, k = i - n * K;
  WT[i] = f2bf(W[(size_t)k * N + n]);
}

// ============ shared GEMM core: 8 waves, WG tile 128x128, wave tile 32x64 ============
// Each wave: 2 (M) x 4 (N) WMMA tiles, K-loop step 32.
// Both A and B fragments are double-buffered in registers: iteration k issues all
// loads for k+32 before running its 8 WMMAs, so WMMAs never wait on same-step loads.
template <typename EPILOGUE>
static __device__ __forceinline__ void gemm_core_128x128(
    const __bf16* __restrict__ A, const __bf16* __restrict__ BT, int K,
    EPILOGUE epi) {
  const int lane = threadIdx.x & 31;
  const int wave = threadIdx.x >> 5;
  const int wm = wave & 3, wn = wave >> 2;           // 4 x 2 wave grid
  const int rowBase = blockIdx.y * 128 + wm * 32;
  const int colBase = blockIdx.x * 128 + wn * 64;

  F8 acc[2][4];
#pragma unroll
  for (int i = 0; i < 2; ++i)
#pragma unroll
    for (int j = 0; j < 4; ++j)
#pragma unroll
      for (int r = 0; r < 8; ++r) acc[i][j].f[r] = 0.0f;

  // preload k=0 fragments
  v16bf a0 = load_a_frag(A, K, rowBase, 0, lane);
  v16bf a1 = load_a_frag(A, K, rowBase + 16, 0, lane);
  v16bf b0 = load_b_frag(BT, K, colBase, 0, lane);
  v16bf b1 = load_b_frag(BT, K, colBase + 16, 0, lane);
  v16bf b2 = load_b_frag(BT, K, colBase + 32, 0, lane);
  v16bf b3 = load_b_frag(BT, K, colBase + 48, 0, lane);

  for (int k0 = 0; k0 < K; k0 += 32) {
    // issue next k-step's loads first (consumed only after this step's WMMAs)
    v16bf a0n = a0, a1n = a1, b0n = b0, b1n = b1, b2n = b2, b3n = b3;
    const int kn = k0 + 32;
    if (kn < K) {
      a0n = load_a_frag(A, K, rowBase, kn, lane);
      a1n = load_a_frag(A, K, rowBase + 16, kn, lane);
      b0n = load_b_frag(BT, K, colBase, kn, lane);
      b1n = load_b_frag(BT, K, colBase + 16, kn, lane);
      b2n = load_b_frag(BT, K, colBase + 32, kn, lane);
      b3n = load_b_frag(BT, K, colBase + 48, kn, lane);
    }
    acc[0][0].v = wmma_bf16(a0, b0, acc[0][0].v);
    acc[1][0].v = wmma_bf16(a1, b0, acc[1][0].v);
    acc[0][1].v = wmma_bf16(a0, b1, acc[0][1].v);
    acc[1][1].v = wmma_bf16(a1, b1, acc[1][1].v);
    acc[0][2].v = wmma_bf16(a0, b2, acc[0][2].v);
    acc[1][2].v = wmma_bf16(a1, b2, acc[1][2].v);
    acc[0][3].v = wmma_bf16(a0, b3, acc[0][3].v);
    acc[1][3].v = wmma_bf16(a1, b3, acc[1][3].v);
    a0 = a0n; a1 = a1n;
    b0 = b0n; b1 = b1n; b2 = b2n; b3 = b3n;
  }

  const int half = lane >> 4, lr = lane & 15;
#pragma unroll
  for (int mt = 0; mt < 2; ++mt)
#pragma unroll
    for (int nt = 0; nt < 4; ++nt)
#pragma unroll
      for (int r = 0; r < 8; ++r) {
        int m = rowBase + mt * 16 + r + 8 * half;
        int n = colBase + nt * 16 + lr;
        epi(m, n, acc[mt][nt].f[r]);
      }
}

// ---------------- QKV projection GEMM ----------------
// C = X(8192x1024) * Wqkv(1024x3072) + b; scatter into Q/K [B,H,S,DK] and V^T [B,H,DK,S]
__global__ __launch_bounds__(256) void gemm_qkv(const __bf16* __restrict__ X,
                                                const __bf16* __restrict__ WT,
                                                const float* __restrict__ bias,
                                                __bf16* __restrict__ qb,
                                                __bf16* __restrict__ kb,
                                                __bf16* __restrict__ vb) {
  gemm_core_128x128(X, WT, 1024, [=](int m, int n, float v) {
    float val = v + bias[n];
    int b = m >> 11, s = m & 2047;
    int sec = n >> 10, nn = n & 1023;
    int h = nn >> 6, d = nn & 63;
    __bf16 bv = f2bf(val);
    if (sec == 0)
      qb[(((size_t)(b * 16 + h) * 2048) + s) * 64 + d] = bv;
    else if (sec == 1)
      kb[(((size_t)(b * 16 + h) * 2048) + s) * 64 + d] = bv;
    else
      vb[(((size_t)(b * 16 + h) * 64) + d) * 2048 + s] = bv;
  });
}

// ---------------- output projection GEMM ----------------
__global__ __launch_bounds__(256) void gemm_head(const __bf16* __restrict__ A,
                                                 const __bf16* __restrict__ WT,
                                                 const float* __restrict__ bias,
                                                 float* __restrict__ out) {
  gemm_core_128x128(A, WT, 1024, [=](int m, int n, float v) {
    out[(size_t)m * 1024 + n] = v + bias[n];
  });
}

// ---------------- causal flash attention ----------------
// grid (S/64, H, B), 128 threads = 4 waves; each wave owns 16 query rows.
__global__ __launch_bounds__(128) void attn_fwd(const __bf16* __restrict__ qg,
                                                const __bf16* __restrict__ kg,
                                                const __bf16* __restrict__ vg,
                                                __bf16* __restrict__ ao) {
  __shared__ __attribute__((aligned(16))) __bf16 lds_p[4][16][64];

  const int lane = threadIdx.x & 31;
  const int wave = threadIdx.x >> 5;
  const int half = lane >> 4, lr = lane & 15;
  const int b = blockIdx.z, h = blockIdx.y;
  const int R0 = blockIdx.x * 64;
  const int qrowBase = R0 + wave * 16;

  const __bf16* Q  = qg + (size_t)(b * 16 + h) * 2048 * 64;
  const __bf16* Km = kg + (size_t)(b * 16 + h) * 2048 * 64;
  const __bf16* Vt = vg + (size_t)(b * 16 + h) * 64 * 2048;

  v16bf aQ[2];
  aQ[0] = load_a_frag(Q, 64, qrowBase, 0, lane);
  aQ[1] = load_a_frag(Q, 64, qrowBase, 32, lane);

  F8 o[4];
  float m_r[8], l_r[8];
#pragma unroll
  for (int dt = 0; dt < 4; ++dt)
#pragma unroll
    for (int r = 0; r < 8; ++r) o[dt].f[r] = 0.0f;
#pragma unroll
  for (int r = 0; r < 8; ++r) { m_r[r] = -1.0e30f; l_r[r] = 0.0f; }

  const int jd = R0 >> 6;  // diagonal key block (uniform across the workgroup)
  for (int j = 0; j <= jd; ++j) {
    // ---- scores: S(16x64) = Q(16x64) @ K^T ----
    F8 sc[4];
#pragma unroll
    for (int nt = 0; nt < 4; ++nt) {
      v8f c = {};
#pragma unroll
      for (int kk = 0; kk < 2; ++kk) {
        v16bf bK = load_b_frag(Km, 64, j * 64 + nt * 16, kk * 32, lane);
        c = wmma_bf16(aQ[kk], bK, c);
      }
      sc[nt].v = c;
    }

    const bool diag = (j == jd);
#pragma unroll
    for (int nt = 0; nt < 4; ++nt)
#pragma unroll
      for (int r = 0; r < 8; ++r) {
        float s = sc[nt].f[r] * 0.125f;  // 1/sqrt(64)
        if (diag) {
          int kcol = j * 64 + nt * 16 + lr;
          int row = qrowBase + r + 8 * half;
          if (kcol > row) s = -1.0e30f;
        }
        sc[nt].f[r] = s;
      }

    // ---- online softmax statistics ----
    float alpha[8];
#pragma unroll
    for (int r = 0; r < 8; ++r) {
      float mx = fmaxf(fmaxf(sc[0].f[r], sc[1].f[r]), fmaxf(sc[2].f[r], sc[3].f[r]));
#pragma unroll
      for (int off = 1; off < 16; off <<= 1)
        mx = fmaxf(mx, __shfl_xor(mx, off, 32));
      float mnew = fmaxf(m_r[r], mx);
      alpha[r] = __expf(m_r[r] - mnew);
      m_r[r] = mnew;
    }
#pragma unroll
    for (int nt = 0; nt < 4; ++nt)
#pragma unroll
      for (int r = 0; r < 8; ++r)
        sc[nt].f[r] = __expf(sc[nt].f[r] - m_r[r]);
#pragma unroll
    for (int r = 0; r < 8; ++r) {
      float sum = sc[0].f[r] + sc[1].f[r] + sc[2].f[r] + sc[3].f[r];
#pragma unroll
      for (int off = 1; off < 16; off <<= 1)
        sum += __shfl_xor(sum, off, 32);
      l_r[r] = l_r[r] * alpha[r] + sum;
    }
#pragma unroll
    for (int dt = 0; dt < 4; ++dt)
#pragma unroll
      for (int r = 0; r < 8; ++r) o[dt].f[r] *= alpha[r];

    // ---- C-layout P -> A-fragment via LDS ----
    __syncthreads();
#pragma unroll
    for (int nt = 0; nt < 4; ++nt)
#pragma unroll
      for (int r = 0; r < 8; ++r)
        lds_p[wave][r + 8 * half][nt * 16 + lr] = f2bf(sc[nt].f[r]);
    __syncthreads();

    // ---- O += P(16x64) @ V(64x64) ----
    const __bf16* pbase = &lds_p[wave][0][0];
#pragma unroll
    for (int kk = 0; kk < 2; ++kk) {
      v16bf aP = load_a_frag(pbase, 64, 0, kk * 32, lane);
#pragma unroll
      for (int dt = 0; dt < 4; ++dt) {
        v16bf bV = load_b_frag(Vt, 2048, dt * 16, j * 64 + kk * 32, lane);
        o[dt].v = wmma_bf16(aP, bV, o[dt].v);
      }
    }
    __syncthreads();
  }

  // ---- finalize: O /= l, write [B,S,D] bf16 ----
#pragma unroll
  for (int r = 0; r < 8; ++r) {
    float inv = 1.0f / l_r[r];
    int srow = qrowBase + r + 8 * half;
#pragma unroll
    for (int dt = 0; dt < 4; ++dt)
      ao[((size_t)(b * 2048 + srow)) * 1024 + h * 64 + dt * 16 + lr] =
          f2bf(o[dt].f[r] * inv);
  }
}

// ---------------- launcher ----------------
extern "C" void kernel_launch(void* const* d_in, const int* in_sizes, int n_in,
                              void* d_out, int out_size, void* d_ws, size_t ws_size,
                              hipStream_t stream) {
  const float* X     = (const float*)d_in[0];
  // d_in[1] = mask (causal handled analytically)
  const float* Wqkv  = (const float*)d_in[2];
  const float* Bqkv  = (const float*)d_in[3];
  const float* Whead = (const float*)d_in[4];
  const float* Bhead = (const float*)d_in[5];
  float* out = (float*)d_out;

  char* ws = (char*)d_ws;
  __bf16* Xbf    = (__bf16*)(ws);                   // 16 MB (8192x1024)
  __bf16* WqkvT  = (__bf16*)(ws + 16777216ull);     //  6 MB (3072x1024)
  __bf16* WheadT = (__bf16*)(ws + 23068672ull);     //  2 MB (1024x1024)
  __bf16* Qbf    = (__bf16*)(ws + 25165824ull);     // 16 MB [B,H,S,DK]
  __bf16* Kbf    = (__bf16*)(ws + 41943040ull);     // 16 MB [B,H,S,DK]
  __bf16* Vbf    = (__bf16*)(ws + 58720256ull);     // 16 MB [B,H,DK,S]
  __bf16* Abf    = (__bf16*)(ws + 75497472ull);     // 16 MB [B,S,D]

  pack_f32_to_bf16<<<32768, 256, 0, stream>>>(X, Xbf, 8192 * 1024);
  pack_transpose<<<12288, 256, 0, stream>>>(Wqkv, WqkvT, 1024, 3072);
  pack_transpose<<<4096, 256, 0, stream>>>(Whead, WheadT, 1024, 1024);

  gemm_qkv<<<dim3(24, 64), 256, 0, stream>>>(Xbf, WqkvT, Bqkv, Qbf, Kbf, Vbf);
  attn_fwd<<<dim3(32, 16, 4), 128, 0, stream>>>(Qbf, Kbf, Vbf, Abf);
  gemm_head<<<dim3(8, 64), 256, 0, stream>>>(Abf, WheadT, Bhead, out);
}
